// crysVQVAE_53145925321458
// MI455X (gfx1250) — matrix operand order repaired and checked by reading, not compiled, not run on hardware
//
#include <hip/hip_runtime.h>
#include <hip/hip_bf16.h>

// ---------------------------------------------------------------------------
// Types for CDNA5 WMMA (wave32): v_wmma_f32_16x16x32_bf16
// ---------------------------------------------------------------------------
typedef __attribute__((ext_vector_type(16))) __bf16 v16bf;
typedef __attribute__((ext_vector_type(8)))  __bf16 v8bf;
typedef __attribute__((ext_vector_type(8)))  float  v8f;

#define DDIM 256              // latent dim
#define KTILE 32              // WMMA K per step (bf16)
#define NKB   (DDIM / KTILE)  // 8 K-steps
#define APITCH 264            // LDS row pitch in halves (528B -> bank-conflict free)

__device__ __forceinline__ unsigned short f2bf(float f) {
  unsigned u = __float_as_uint(f);
  unsigned r = u + 0x7FFFu + ((u >> 16) & 1u);   // round-to-nearest-even
  return (unsigned short)(r >> 16);
}

__device__ __forceinline__ v8f vzero8() {
  v8f z = {0.f, 0.f, 0.f, 0.f, 0.f, 0.f, 0.f, 0.f};
  return z;
}

__device__ __forceinline__ float waveReduceSum(float v) {
#pragma unroll
  for (int off = 16; off > 0; off >>= 1) v += __shfl_xor(v, off, 32);
  return v;
}

// Load an A fragment (16x32 bf16, ISA layout: K = j + 8*half + (j>=8?8:0))
// from a row-major bf16 LDS tile with pitch APITCH; rowOff selects which
// 16-row band of the staged tile this fragment covers.
__device__ __forceinline__ v16bf loadAfrag(const unsigned short* As, int lane,
                                           int kb, int rowOff) {
  const int half = lane >> 4;
  const int mrow = (lane & 15) + rowOff;
  const int a1 = mrow * APITCH + kb * KTILE + 8 * half;
  v8bf lo = *(const v8bf*)(&As[a1]);        // K = k0 + 8h .. +7
  v8bf hi = *(const v8bf*)(&As[a1 + 16]);   // K = k0 + 16 + 8h .. +7
  return __builtin_shufflevector(lo, hi, 0, 1, 2, 3, 4, 5, 6, 7,
                                 8, 9, 10, 11, 12, 13, 14, 15);
}

// B fragments are pre-packed: frag (kt,nt) = 32 lanes x 16 halves contiguous.
__device__ __forceinline__ v16bf loadBfrag(const unsigned short* Wp, int ntilesN,
                                           int kb, int nt, int lane) {
  return *(const v16bf*)(Wp + (((size_t)(kb * ntilesN + nt) * 32 + lane) << 4));
}

// ---------------------------------------------------------------------------
// Pack f32 weight [256, Nsrc] (row-major, K-major rows) into bf16 B-fragment
// layout, N padded to Npad (multiple of 16), zeros in the pad.
// B layout (ISA 7.12.5 pattern, 32x16): lane l -> n = nt*16+(l&15),
// element j -> K = kt*32 + 8*(l>>4) + (j&7) + (j>=8?16:0).
// ---------------------------------------------------------------------------
__global__ __launch_bounds__(256) void pack_weight_kernel(
    const float* __restrict__ W, unsigned short* __restrict__ out,
    int Nsrc, int Npad) {
  int gid = blockIdx.x * 256 + threadIdx.x;
  int total = DDIM * Npad;
  if (gid >= total) return;
  int tile = gid >> 9;          // 512 halves per fragment
  int within = gid & 511;
  int lane = within >> 4;
  int j = within & 15;
  int ntilesN = Npad >> 4;
  int kt = tile / ntilesN;
  int nt = tile - kt * ntilesN;
  int k = kt * 32 + ((lane >> 4) << 3) + (j & 7) + ((j >> 3) << 4);
  int n = nt * 16 + (lane & 15);
  float v = (n < Nsrc) ? W[k * Nsrc + n] : 0.f;
  out[gid] = f2bf(v);
}

// ---------------------------------------------------------------------------
// VQ: wave-per-row argmin over 100 codes, write z2q = emb[best], accumulate
// sum((quant - z2)^2).  d_k ~ |e_k|^2 - 2 z2.e_k  (|z2|^2 constant per row).
// ---------------------------------------------------------------------------
__global__ __launch_bounds__(256) void vq_kernel(
    const float* __restrict__ z2, const float* __restrict__ emb,
    float* __restrict__ z2q, float* __restrict__ vqAcc) {
  const int lane = threadIdx.x & 31;
  const int wave = threadIdx.x >> 5;
  const int row = blockIdx.x * 8 + wave;
  const float* x = z2 + (size_t)row * DDIM;
  float xr[8];
#pragma unroll
  for (int t = 0; t < 8; ++t) xr[t] = x[lane + 32 * t];
  float bestd = 3.4e38f;
  int bestk = 0;
  for (int k = 0; k < 100; ++k) {
    const float* e = emb + k * DDIM;
    float s = 0.f, q = 0.f;
#pragma unroll
    for (int t = 0; t < 8; ++t) {
      float ev = e[lane + 32 * t];
      s = fmaf(xr[t], ev, s);
      q = fmaf(ev, ev, q);
    }
    float d = q - 2.f * s;
#pragma unroll
    for (int off = 16; off > 0; off >>= 1) d += __shfl_xor(d, off, 32);
    if (d < bestd) { bestd = d; bestk = k; }
  }
  const float* e = emb + bestk * DDIM;
  float ls = 0.f;
#pragma unroll
  for (int t = 0; t < 8; ++t) {
    float ev = e[lane + 32 * t];
    z2q[(size_t)row * DDIM + lane + 32 * t] = ev;
    float df = ev - xr[t];
    ls = fmaf(df, df, ls);
  }
  ls = waveReduceSum(ls);
  if (lane == 0) atomicAdd(vqAcc, ls);
}

// ---------------------------------------------------------------------------
// GEMM1: H = X @ W1 + b1.  128 threads = 4 waves; block does 32 rows x 256
// cols (wave owns a 32x64 strip -> each B fragment feeds 2 wmma).
// ---------------------------------------------------------------------------
__global__ __launch_bounds__(128) void gemm1_kernel(
    const float* __restrict__ X, const unsigned short* __restrict__ Wp,
    const float* __restrict__ bias, float* __restrict__ H) {
  __shared__ unsigned short As[32 * APITCH];
  const int tid = threadIdx.x;
  const size_t rowbase = (size_t)blockIdx.x * 32;
  const float* src = X + rowbase * DDIM;
#pragma unroll
  for (int i = 0; i < 64; ++i) {
    int idx = i * 128 + tid;
    As[(idx >> 8) * APITCH + (idx & 255)] = f2bf(src[idx]);
  }
  __syncthreads();
  const int lane = tid & 31;
  const int wave = tid >> 5;
  const int half = lane >> 4;
  const int nl = lane & 15;
  v8f accA[4] = {vzero8(), vzero8(), vzero8(), vzero8()};  // rows 0..15
  v8f accB[4] = {vzero8(), vzero8(), vzero8(), vzero8()};  // rows 16..31
  for (int kb = 0; kb < NKB; ++kb) {
    v16bf afA = loadAfrag(As, lane, kb, 0);
    v16bf afB = loadAfrag(As, lane, kb, 16);
#pragma unroll
    for (int n = 0; n < 4; ++n) {
      v16bf bf = loadBfrag(Wp, DDIM / 16, kb, wave * 4 + n, lane);
      accA[n] = __builtin_amdgcn_wmma_f32_16x16x32_bf16(
          false, afA, false, bf, (short)0, accA[n], false, false);
      accB[n] = __builtin_amdgcn_wmma_f32_16x16x32_bf16(
          false, afB, false, bf, (short)0, accB[n], false, false);
    }
  }
#pragma unroll
  for (int n = 0; n < 4; ++n) {
    int col = (wave * 4 + n) * 16 + nl;
    float bv = bias[col];
#pragma unroll
    for (int r = 0; r < 8; ++r) {
      size_t row = rowbase + r + 8 * half;
      H[row * DDIM + col] = accA[n][r] + bv;
      H[(row + 16) * DDIM + col] = accB[n][r] + bv;
    }
  }
}

// ---------------------------------------------------------------------------
// BatchNorm column stats (sum / sumsq) over B rows, then fold into per-column
// affine a,c so GEMM2 can fuse normalize+relu into its A-load.
// ---------------------------------------------------------------------------
__global__ __launch_bounds__(256) void bn_stats_kernel(
    const float* __restrict__ H, float* __restrict__ musum,
    float* __restrict__ sqsum, int rowsPerBlock) {
  const int col = threadIdx.x;
  const size_t r0 = (size_t)blockIdx.x * rowsPerBlock;
  float s = 0.f, q = 0.f;
  for (int i = 0; i < rowsPerBlock; ++i) {
    float v = H[(r0 + i) * DDIM + col];
    s += v;
    q = fmaf(v, v, q);
  }
  atomicAdd(&musum[col], s);
  atomicAdd(&sqsum[col], q);
}

__global__ void bn_fin_kernel(const float* __restrict__ musum,
                              const float* __restrict__ sqsum,
                              const float* __restrict__ g,
                              const float* __restrict__ beta,
                              float* __restrict__ aArr, float* __restrict__ cArr,
                              float invB) {
  int t = threadIdx.x;
  float mu = musum[t] * invB;
  float var = sqsum[t] * invB - mu * mu;
  float aa = rsqrtf(var + 1e-5f) * g[t];
  aArr[t] = aa;
  cArr[t] = beta[t] - mu * aa;
}

// ---------------------------------------------------------------------------
// GEMM2 fused: Y = relu(H*a + c) @ W2 + b2 ; accumulate sum((Y-target)^2);
// optionally write Y (needed for p1).  32 rows x 256 cols per block.
// ---------------------------------------------------------------------------
__global__ __launch_bounds__(128) void gemm2_kernel(
    const float* __restrict__ H, const float* __restrict__ aScale,
    const float* __restrict__ cShift, const unsigned short* __restrict__ Wp,
    const float* __restrict__ bias, const float* __restrict__ target,
    float* __restrict__ Yout, int writeY, float* __restrict__ lossAcc) {
  __shared__ unsigned short As[32 * APITCH];
  const int tid = threadIdx.x;
  const size_t rowbase = (size_t)blockIdx.x * 32;
  const float* src = H + rowbase * DDIM;
#pragma unroll
  for (int i = 0; i < 64; ++i) {
    int idx = i * 128 + tid;
    int c = idx & 255;
    float v = fmaf(src[idx], aScale[c], cShift[c]);
    v = v > 0.f ? v : 0.f;                    // ReLU fused into A staging
    As[(idx >> 8) * APITCH + c] = f2bf(v);
  }
  __syncthreads();
  const int lane = tid & 31;
  const int wave = tid >> 5;
  const int half = lane >> 4;
  const int nl = lane & 15;
  v8f accA[4] = {vzero8(), vzero8(), vzero8(), vzero8()};
  v8f accB[4] = {vzero8(), vzero8(), vzero8(), vzero8()};
  for (int kb = 0; kb < NKB; ++kb) {
    v16bf afA = loadAfrag(As, lane, kb, 0);
    v16bf afB = loadAfrag(As, lane, kb, 16);
#pragma unroll
    for (int n = 0; n < 4; ++n) {
      v16bf bf = loadBfrag(Wp, DDIM / 16, kb, wave * 4 + n, lane);
      accA[n] = __builtin_amdgcn_wmma_f32_16x16x32_bf16(
          false, afA, false, bf, (short)0, accA[n], false, false);
      accB[n] = __builtin_amdgcn_wmma_f32_16x16x32_bf16(
          false, afB, false, bf, (short)0, accB[n], false, false);
    }
  }
  float ls = 0.f;
#pragma unroll
  for (int n = 0; n < 4; ++n) {
    int col = (wave * 4 + n) * 16 + nl;
    float bv = bias[col];
#pragma unroll
    for (int r = 0; r < 8; ++r) {
      size_t row = rowbase + r + 8 * half;
      float y0 = accA[n][r] + bv;
      float y1 = accB[n][r] + bv;
      float d0 = y0 - target[row * DDIM + col];
      float d1 = y1 - target[(row + 16) * DDIM + col];
      ls = fmaf(d0, d0, ls);
      ls = fmaf(d1, d1, ls);
      if (writeY) {
        Yout[row * DDIM + col] = y0;
        Yout[(row + 16) * DDIM + col] = y1;
      }
    }
  }
  ls = waveReduceSum(ls);
  if (lane == 0) atomicAdd(lossAcc, ls);
}

// ---------------------------------------------------------------------------
// Atom head: logits = zn @ atom_W + b (N cols padded 100->112 = 7 tiles),
// fused masked log-softmax + CE + segment atomics.  1 wave per 16 rows.
// HBM-bound on the zn stream -> prefetch the next block's tile.
// ---------------------------------------------------------------------------
__global__ __launch_bounds__(32) void atom_kernel(
    const float* __restrict__ zn, const unsigned short* __restrict__ Wp,
    const float* __restrict__ atom_b, const int* __restrict__ atomic_nums,
    const int* __restrict__ batch_idx, float* __restrict__ sums,
    float* __restrict__ counts, int nrows) {
  __shared__ unsigned short As[16 * APITCH];
  const int lane = threadIdx.x;
  const size_t rowbase = (size_t)blockIdx.x * 16;
  const float* src = zn + rowbase * DDIM;
  if (rowbase + 16 < (size_t)nrows)  // stream hint for the next tile
    __builtin_prefetch(src + 16 * DDIM + lane * 32, 0, 1);
#pragma unroll
  for (int i = 0; i < 128; ++i) {
    int idx = i * 32 + lane;
    As[(idx >> 8) * APITCH + (idx & 255)] = f2bf(src[idx]);
  }
  __syncthreads();
  const int half = lane >> 4;
  const int nl = lane & 15;
  v8f acc[7] = {vzero8(), vzero8(), vzero8(), vzero8(),
                vzero8(), vzero8(), vzero8()};
  for (int kb = 0; kb < NKB; ++kb) {
    v16bf af = loadAfrag(As, lane, kb, 0);
#pragma unroll
    for (int nt = 0; nt < 7; ++nt) {
      v16bf bf = loadBfrag(Wp, 7, kb, nt, lane);
      acc[nt] = __builtin_amdgcn_wmma_f32_16x16x32_bf16(
          false, af, false, bf, (short)0, acc[nt], false, false);
    }
  }
  float bb[7];
#pragma unroll
  for (int nt = 0; nt < 7; ++nt) {
    int col = nt * 16 + nl;
    bb[nt] = (col < 100) ? atom_b[col] : -1e30f;   // mask padded cols
  }
#pragma unroll
  for (int r = 0; r < 8; ++r) {
    size_t row = rowbase + r + 8 * half;       // each half-wave owns 8 rows
    float lg[7];
    float mx = -1e30f;
#pragma unroll
    for (int nt = 0; nt < 7; ++nt) {
      lg[nt] = acc[nt][r] + bb[nt];
      mx = fmaxf(mx, lg[nt]);
    }
#pragma unroll
    for (int off = 8; off > 0; off >>= 1) mx = fmaxf(mx, __shfl_xor(mx, off, 32));
    float se = 0.f;
#pragma unroll
    for (int nt = 0; nt < 7; ++nt) se += __expf(lg[nt] - mx);
#pragma unroll
    for (int off = 8; off > 0; off >>= 1) se += __shfl_xor(se, off, 32);
    float lse = mx + __logf(se);
    int t = atomic_nums[row];
    float tl = 0.f;
#pragma unroll
    for (int nt = 0; nt < 7; ++nt) {
      int col = nt * 16 + nl;
      tl += (col == t) ? acc[nt][r] : 0.f;
    }
#pragma unroll
    for (int off = 8; off > 0; off >>= 1) tl += __shfl_xor(tl, off, 32);
    float ce = lse - (tl + atom_b[t]);
    if (nl == 0) {
      int b = batch_idx[row];
      atomicAdd(&sums[b], ce);
      atomicAdd(&counts[b], 1.f);
    }
  }
}

__global__ __launch_bounds__(256) void atom_reduce_kernel(
    const float* __restrict__ sums, const float* __restrict__ counts,
    float* __restrict__ acc, int B) {
  int i = blockIdx.x * 256 + threadIdx.x;
  float v = (i < B) ? sums[i] / fmaxf(counts[i], 1.f) : 0.f;
  v = waveReduceSum(v);
  if ((threadIdx.x & 31) == 0) atomicAdd(acc, v);
}

// ---------------------------------------------------------------------------
// Lattice head: wave-per-row, 6 dot products, MSE accumulation.
// ---------------------------------------------------------------------------
__global__ __launch_bounds__(256) void latt_kernel(
    const float* __restrict__ z2q, const float* __restrict__ W,
    const float* __restrict__ b6, const float* __restrict__ lat,
    const float* __restrict__ mean, const float* __restrict__ stdv,
    float* __restrict__ acc) {
  const int lane = threadIdx.x & 31;
  const int wave = threadIdx.x >> 5;
  const size_t row = (size_t)blockIdx.x * 8 + wave;
  const float* x = z2q + row * DDIM;
  float a[6] = {0.f, 0.f, 0.f, 0.f, 0.f, 0.f};
#pragma unroll
  for (int t = 0; t < 8; ++t) {
    int d = lane + 32 * t;
    float xv = x[d];
#pragma unroll
    for (int j = 0; j < 6; ++j) a[j] = fmaf(xv, W[d * 6 + j], a[j]);
  }
  float s = 0.f;
#pragma unroll
  for (int j = 0; j < 6; ++j) {
    float v = a[j];
#pragma unroll
    for (int off = 16; off > 0; off >>= 1) v += __shfl_xor(v, off, 32);
    if (lane == 0) {
      float pred = v + b6[j];
      float tgt = (lat[row * 6 + j] - mean[j]) / stdv[j];
      float df = pred - tgt;
      s = fmaf(df, df, s);
    }
  }
  if (lane == 0) atomicAdd(acc, s);
}

// ---------------------------------------------------------------------------
// Finalize: combine accumulators into the 5 f32 outputs.
// sc = [vqSum, pSum, iSum, cSum, lattSum, atomSum]
// ---------------------------------------------------------------------------
__global__ void finalize_kernel(const float* __restrict__ sc,
                                float* __restrict__ out, float invBD,
                                float invB6, float invB) {
  float vq = 2.f * sc[0] * invBD;
  float cyc = (sc[1] + sc[2] + sc[3]) * invBD;
  float latt = sc[4] * invB6 * 10.f;
  float atom = sc[5] * invB;
  out[0] = cyc + atom + latt + vq;
  out[1] = cyc;
  out[2] = atom;
  out[3] = latt;
  out[4] = vq;
}

// ---------------------------------------------------------------------------
extern "C" void kernel_launch(void* const* d_in, const int* in_sizes, int n_in,
                              void* d_out, int out_size, void* d_ws,
                              size_t ws_size, hipStream_t stream) {
  const float* z1 = (const float*)d_in[0];
  const float* z2 = (const float*)d_in[1];
  const float* zn = (const float*)d_in[2];
  const float* emb = (const float*)d_in[3];
  const float* proj_W1 = (const float*)d_in[4];
  const float* proj_b1 = (const float*)d_in[5];
  const float* proj_g = (const float*)d_in[6];
  const float* proj_beta = (const float*)d_in[7];
  const float* proj_W2 = (const float*)d_in[8];
  const float* proj_b2 = (const float*)d_in[9];
  const float* invp_W1 = (const float*)d_in[10];
  const float* invp_b1 = (const float*)d_in[11];
  const float* invp_g = (const float*)d_in[12];
  const float* invp_beta = (const float*)d_in[13];
  const float* invp_W2 = (const float*)d_in[14];
  const float* invp_b2 = (const float*)d_in[15];
  const float* latt_W = (const float*)d_in[16];
  const float* latt_b = (const float*)d_in[17];
  const float* atom_W = (const float*)d_in[18];
  const float* atom_b = (const float*)d_in[19];
  const float* sc_mean = (const float*)d_in[20];
  const float* sc_std = (const float*)d_in[21];
  const float* lsc_lat = (const float*)d_in[22];
  const int* batch_idx = (const int*)d_in[24];
  const int* atomic_nums = (const int*)d_in[25];
  (void)n_in; (void)out_size; (void)ws_size;

  const int B = in_sizes[0] / DDIM;  // 32768
  const int N = in_sizes[24];        // 262144

  // ---- workspace carve-up (256B aligned) ----
  char* ws = (char*)d_ws;
  size_t cur = 0;
  auto alloc = [&](size_t bytes) -> void* {
    void* p = ws + cur;
    cur += (bytes + 255) & ~(size_t)255;
    return p;
  };
  float* scal = (float*)alloc(256);           // 6 loss accumulators
  float* musum = (float*)alloc(DDIM * 4);
  float* sqsum = (float*)alloc(DDIM * 4);
  float* aArr = (float*)alloc(DDIM * 4);
  float* cArr = (float*)alloc(DDIM * 4);
  float* segSums = (float*)alloc((size_t)B * 4);
  float* segCnts = (float*)alloc((size_t)B * 4);
  unsigned short* wpP1 = (unsigned short*)alloc((size_t)DDIM * DDIM * 2);
  unsigned short* wpP2 = (unsigned short*)alloc((size_t)DDIM * DDIM * 2);
  unsigned short* wpI1 = (unsigned short*)alloc((size_t)DDIM * DDIM * 2);
  unsigned short* wpI2 = (unsigned short*)alloc((size_t)DDIM * DDIM * 2);
  unsigned short* wpAt = (unsigned short*)alloc((size_t)DDIM * 112 * 2);
  float* z2q = (float*)alloc((size_t)B * DDIM * 4);
  float* p1 = (float*)alloc((size_t)B * DDIM * 4);
  float* hbuf = (float*)alloc((size_t)B * DDIM * 4);

  // ---- zero accumulators (poisoned 0xAA by harness, must re-init per call) --
  hipMemsetAsync(scal, 0, 256, stream);
  hipMemsetAsync(segSums, 0, (size_t)B * 4, stream);
  hipMemsetAsync(segCnts, 0, (size_t)B * 4, stream);

  // ---- pack weights into B-fragment layout (bf16) ----
  const int packBlocksDD = (DDIM * DDIM + 255) / 256;
  pack_weight_kernel<<<packBlocksDD, 256, 0, stream>>>(proj_W1, wpP1, DDIM, DDIM);
  pack_weight_kernel<<<packBlocksDD, 256, 0, stream>>>(proj_W2, wpP2, DDIM, DDIM);
  pack_weight_kernel<<<packBlocksDD, 256, 0, stream>>>(invp_W1, wpI1, DDIM, DDIM);
  pack_weight_kernel<<<packBlocksDD, 256, 0, stream>>>(invp_W2, wpI2, DDIM, DDIM);
  pack_weight_kernel<<<(DDIM * 112 + 255) / 256, 256, 0, stream>>>(atom_W, wpAt,
                                                                   100, 112);

  // ---- VQ: z2q = emb[argmin], vq accumulation ----
  vq_kernel<<<B / 8, 256, 0, stream>>>(z2, emb, z2q, &scal[0]);

  // ---- MLP-BN passes ----
  auto mlp_pass = [&](const float* X, const unsigned short* W1p,
                      const float* b1, const float* g, const float* be,
                      const unsigned short* W2p, const float* b2,
                      const float* tgt, float* yout, int writeY,
                      float* lossSlot) {
    gemm1_kernel<<<B / 32, 128, 0, stream>>>(X, W1p, b1, hbuf);
    hipMemsetAsync(musum, 0, DDIM * 4, stream);
    hipMemsetAsync(sqsum, 0, DDIM * 4, stream);
    bn_stats_kernel<<<64, 256, 0, stream>>>(hbuf, musum, sqsum, B / 64);
    bn_fin_kernel<<<1, 256, 0, stream>>>(musum, sqsum, g, be, aArr, cArr,
                                         1.f / (float)B);
    gemm2_kernel<<<B / 32, 128, 0, stream>>>(hbuf, aArr, cArr, W2p, b2, tgt,
                                             yout, writeY, lossSlot);
  };
  // pass P: p1 = mlp(z1, proj); loss_p vs z2q; keep p1
  mlp_pass(z1, wpP1, proj_b1, proj_g, proj_beta, wpP2, proj_b2, z2q, p1, 1,
           &scal[1]);
  // pass I: mlp(z2q, invp); loss_i vs z1
  mlp_pass(z2q, wpI1, invp_b1, invp_g, invp_beta, wpI2, invp_b2, z1, p1, 0,
           &scal[2]);
  // pass C: mlp(p1, invp); loss_c vs z1
  mlp_pass(p1, wpI1, invp_b1, invp_g, invp_beta, wpI2, invp_b2, z1, p1, 0,
           &scal[3]);

  // ---- lattice loss ----
  latt_kernel<<<B / 8, 256, 0, stream>>>(z2q, latt_W, latt_b, lsc_lat, sc_mean,
                                         sc_std, &scal[4]);

  // ---- atom CE head ----
  atom_kernel<<<N / 16, 32, 0, stream>>>(zn, wpAt, atom_b, atomic_nums,
                                         batch_idx, segSums, segCnts, N);
  atom_reduce_kernel<<<(B + 255) / 256, 256, 0, stream>>>(segSums, segCnts,
                                                          &scal[5], B);

  // ---- finalize ----
  finalize_kernel<<<1, 1, 0, stream>>>(scal, (float*)d_out,
                                       1.f / ((float)B * DDIM),
                                       1.f / ((float)B * 6.f),
                                       1.f / (float)B);
}
